// SUBJ_DKVMN_74998718923402
// MI455X (gfx1250) — compile-verified, matrix operand order checked by reading, not run on hardware
//
#include <hip/hip_runtime.h>
#include <hip/hip_bf16.h>
#include <math.h>

// ---------------- problem constants (match reference) ----------------
static constexpr int NUM_Q = 10000;
static constexpr int Bsz   = 64;
static constexpr int Nn    = 200;
static constexpr int Dd    = 200;   // DIM_S
static constexpr int BN    = Bsz * Nn;      // 12800 rows, multiple of 16
static constexpr int SM    = 64;    // SIZE_M
static constexpr int BD    = 768;   // BERT_D
static constexpr int NH    = 5;
static constexpr int HD    = 40;

// padded dims for WMMA (K multiple of 32; weight rows padded to multiple of 64
// so every wave in a 4-wave block owns a valid 16-col tile -> no divergence
// around __syncthreads / WMMA, EXEC stays all-ones)
static constexpr int KP200 = 224;
static constexpr int KP400 = 416;
static constexpr int KP768 = 768;
static constexpr int NP200 = 256;
static constexpr int KPMAX = 768;

typedef __attribute__((ext_vector_type(16))) _Float16 v16h;
typedef __attribute__((ext_vector_type(8)))  _Float16 v8h;
typedef __attribute__((ext_vector_type(8)))  float    v8f;
typedef __attribute__((ext_vector_type(4)))  unsigned int v4u;
typedef __attribute__((ext_vector_type(8)))  int      v8i;
typedef __attribute__((ext_vector_type(4)))  int      v4i;

#if __has_builtin(__builtin_amdgcn_tensor_load_to_lds) && __has_builtin(__builtin_amdgcn_s_wait_tensorcnt)
#define USE_TDM 1
#endif

// ---------------- fragment load (16-bit A layout, ISA 7.12.2) ----------------
// lane<16 : elements 0..7 = K base+0..7 , 8..15 = K base+16..23
// lane>=16: base is +8 (K 8..15 / 24..31). B mirrors A over weight rows since
// C = A * W^T  =>  B[k,n] = W[n,k], per-lane contiguous K run of row n.
__device__ __forceinline__ v16h load_frag_g(const _Float16* __restrict__ p) {
  v8h lo = *(const v8h*)(p);
  v8h hi = *(const v8h*)(p + 16);
  v16h f;
#pragma unroll
  for (int i = 0; i < 8; ++i) { f[i] = lo[i]; f[8 + i] = hi[i]; }
  return f;
}
__device__ __forceinline__ v16h load_frag_lds(const _Float16* p) {
  v8h lo = *(const v8h*)(p);
  v8h hi = *(const v8h*)(p + 16);
  v16h f;
#pragma unroll
  for (int i = 0; i < 8; ++i) { f[i] = lo[i]; f[8 + i] = hi[i]; }
  return f;
}

// ---------------- convert f32 -> f16 with zero padding ----------------
__global__ void cvt_pad_kernel(const float* __restrict__ src, _Float16* __restrict__ dst,
                               int rows, int K, int rowsPad, int Kpad) {
  int idx = blockIdx.x * blockDim.x + threadIdx.x;
  int total = rowsPad * Kpad;
  if (idx >= total) return;
  int r = idx / Kpad, c = idx - r * Kpad;
  float v = (r < rows && c < K) ? src[(size_t)r * K + c] : 0.f;
  dst[idx] = (_Float16)v;
}

// concat([s0, s1], -1) -> padded f16
__global__ void cvt_pad2_kernel(const float* __restrict__ s0, const float* __restrict__ s1,
                                _Float16* __restrict__ dst, int rows, int K0, int K1, int Kpad) {
  int idx = blockIdx.x * blockDim.x + threadIdx.x;
  int total = rows * Kpad;
  if (idx >= total) return;
  int r = idx / Kpad, c = idx - r * Kpad;
  float v = 0.f;
  if (c < K0)            v = s0[(size_t)r * K0 + c];
  else if (c < K0 + K1)  v = s1[(size_t)r * K1 + (c - K0)];
  dst[idx] = (_Float16)v;
}

// ---------------- embedding gathers ----------------
__global__ void gather_kernel(const float* __restrict__ kemb, const float* __restrict__ vemb,
                              const int* __restrict__ q, const int* __restrict__ r,
                              float* __restrict__ kout, float* __restrict__ vout) {
  int idx = blockIdx.x * blockDim.x + threadIdx.x;
  if (idx >= BN * Dd) return;
  int bn = idx / Dd, d = idx - bn * Dd;
  int qi = q[bn];
  kout[idx] = kemb[(size_t)qi * Dd + d];
  int xi = qi + r[bn] * NUM_Q;
  vout[idx] = vemb[(size_t)xi * Dd + d];
}

// ---------------- WMMA GEMM: C[M,Nc] = act(A[M,Kpad] * W[NcPad,Kpad]^T + bias [+P]) ----------------
// A-tile (16 rows x Kpad f16) staged once per block into LDS via the Tensor
// Data Mover (tensor_load_to_lds + s_wait_tensorcnt), shared by all 4 waves.
// act: 0=none 1=sigmoid 2=tanh 3=relu.  Padd: optional [Nmod, Nc] broadcast add.
__global__ __launch_bounds__(128) void gemm_wmma_kernel(
    const _Float16* __restrict__ A, const _Float16* __restrict__ W,
    const float* __restrict__ bias, const float* __restrict__ Padd,
    float* __restrict__ C, int M, int Nc, int NcPad, int Kpad, int act, int Nmod) {
  __shared__ _Float16 Atile[16 * KPMAX];   // sole LDS object -> LDS offset 0
  int lane = threadIdx.x & 31;
  int wave = threadIdx.x >> 5;
  int row0 = blockIdx.x * 16;
  int col0 = (blockIdx.y * 4 + wave) * 16;

#ifdef USE_TDM
  if (threadIdx.x < 32) {   // one wave issues the TDM descriptor (EXEC ignored by TDM)
    unsigned long long ga = (unsigned long long)(A + (size_t)row0 * Kpad);
    // D# group0: count=1 | lds_addr | 57b global addr | type=2 ("image")
    v4u g0;
    g0[0] = 1u;                                   // count=1, is_restore=0
    g0[1] = 0u;                                   // lds_addr = offset of Atile
    g0[2] = (unsigned int)(ga & 0xFFFFFFFFu);
    g0[3] = (unsigned int)((ga >> 32) & 0x01FFFFFFu) | (2u << 30);
    // D# group1: data_size=2B; tensor = Kpad x 16 rows; tile = Kpad x 16;
    // tensor_dim0_stride = Kpad (elements)
    v8i g1;
    g1[0] = (1 << 16);                            // workgroup_mask=0, data_size=1 (2B)
    g1[1] = (Kpad & 0xFFFF) << 16;                // tensor_dim0 low16
    g1[2] = ((Kpad >> 16) & 0xFFFF) | (16 << 16); // tensor_dim0 hi | tensor_dim1=16
    g1[3] = (Kpad & 0xFFFF) << 16;                // tensor_dim1 hi=0 | tile_dim0=Kpad
    g1[4] = 16;                                   // tile_dim1=16, tile_dim2=0
    g1[5] = Kpad;                                 // tensor_dim0_stride low32
    g1[6] = 0;
    g1[7] = 0;
    v4i gz = {0, 0, 0, 0};
#if defined(__clang_major__) && __clang_major__ >= 23
    v8i gz8 = {0, 0, 0, 0, 0, 0, 0, 0};
    __builtin_amdgcn_tensor_load_to_lds(g0, g1, gz, gz, gz8, 0);
#else
    __builtin_amdgcn_tensor_load_to_lds(g0, g1, gz, gz, 0);
#endif
    __builtin_amdgcn_s_wait_tensorcnt(0);
  }
  // The TDM builtin names the LDS destination only inside the descriptor, so
  // the compiler sees no store to Atile and would fold the LDS reads to undef
  // (observed: A-fragment collapsed onto the accumulator + 5 v_nop hazards).
  // Escape the pointer through an asm memory clobber so the ds_loads are real.
  {
    const _Float16* esc = &Atile[0];
    asm volatile("" : : "v"(esc) : "memory");
  }
#else
  // fallback: cooperative global->LDS copy of the A tile
  for (int i = threadIdx.x; i < 16 * (Kpad / 8); i += 128) {
    int rr = i / (Kpad / 8), cc = (i - rr * (Kpad / 8)) * 8;
    *(v8h*)&Atile[rr * Kpad + cc] = *(const v8h*)&A[(size_t)(row0 + rr) * Kpad + cc];
  }
#endif
  __syncthreads();

  int l16  = lane & 15;
  int koff = (lane & 16) ? 8 : 0;
  const _Float16* ap = Atile + l16 * Kpad + koff;
  const _Float16* wp = W + (size_t)(col0 + l16) * Kpad + koff;  // col0+15 < NcPad always
  v8f c = {};
  for (int kk = 0; kk < Kpad; kk += 32) {
    v16h af = load_frag_lds(ap + kk);
    v16h bf = load_frag_g(wp + kk);
    c = __builtin_amdgcn_wmma_f32_16x16x32_f16(false, af, false, bf, (short)0, c, false, false);
  }
  int ncol = col0 + l16;
  if (ncol >= Nc) return;                 // divergence only after all WMMAs/barriers
  int mbase = row0 + ((lane & 16) ? 8 : 0);
  float bv = bias ? bias[ncol] : 0.f;
#pragma unroll
  for (int r2 = 0; r2 < 8; ++r2) {
    int mrow = mbase + r2;
    float x = c[r2] + bv;
    if (Padd) x += Padd[(size_t)(mrow % Nmod) * Nc + ncol];
    if (act == 1)      x = 1.f / (1.f + expf(-x));
    else if (act == 2) x = tanhf(x);
    else if (act == 3) x = fmaxf(x, 0.f);
    C[(size_t)mrow * Nc + ncol] = x;
  }
}

// ---------------- row softmax (w over SIZE_M) ----------------
__global__ void softmax_rows_kernel(float* __restrict__ x, int rows, int width) {
  int rI = blockIdx.x * blockDim.x + threadIdx.x;
  if (rI >= rows) return;
  float* p = x + (size_t)rI * width;
  float mx = -1e30f;
  for (int i = 0; i < width; ++i) mx = fmaxf(mx, p[i]);
  float s = 0.f;
  for (int i = 0; i < width; ++i) { float e = expf(p[i] - mx); p[i] = e; s += e; }
  float inv = 1.f / s;
  for (int i = 0; i < width; ++i) p[i] *= inv;
}

// ---------------- sequential erase-add memory recurrence ----------------
// one block per batch; 64x200 f32 state resident in LDS (51.2KB of 320KB/WGP).
// computes read_t = w_t . Mv_t BEFORE the update, streams every Mv slice to d_out.
__global__ __launch_bounds__(256) void recurrence_kernel(
    const float* __restrict__ Mv0, const float* __restrict__ w,
    const float* __restrict__ e, const float* __restrict__ a,
    float* __restrict__ readout, float* __restrict__ outMv) {
  __shared__ float Mv[SM * Dd];
  __shared__ float wv[SM];
  __shared__ float ev[Dd];
  __shared__ float av[Dd];
  int b = blockIdx.x, tid = threadIdx.x;
  float* slice0 = outMv + (size_t)b * (Nn + 1) * SM * Dd;
  for (int i = tid; i < SM * Dd; i += 256) { float v0 = Mv0[i]; Mv[i] = v0; slice0[i] = v0; }
  __syncthreads();
  for (int t = 0; t < Nn; ++t) {
    size_t bn = (size_t)b * Nn + t;
    if (tid < SM) wv[tid] = w[bn * SM + tid];
    if (tid < Dd) { ev[tid] = e[bn * Dd + tid]; av[tid] = a[bn * Dd + tid]; }
    __syncthreads();
    if (tid < Dd) {
      float acc = 0.f;
#pragma unroll 4
      for (int m = 0; m < SM; ++m) acc += wv[m] * Mv[m * Dd + tid];
      readout[bn * Dd + tid] = acc;
    }
    __syncthreads();
    float* sliceT = outMv + ((size_t)b * (Nn + 1) + (t + 1)) * SM * Dd;
    for (int i = tid; i < SM * Dd; i += 256) {
      int m = i / Dd, d = i - m * Dd;
      float val = Mv[i] * (1.f - wv[m] * ev[d]) + wv[m] * av[d];
      Mv[i] = val;
      sliceT[i] = val;
    }
    if (t + 1 < Nn) {           // gfx1250 global_prefetch of next step's inputs
      __builtin_prefetch(&w[(bn + 1) * SM], 0, 0);
      __builtin_prefetch(&e[(bn + 1) * Dd], 0, 0);
      __builtin_prefetch(&a[(bn + 1) * Dd], 0, 0);
    }
    __syncthreads();
  }
}

// ---------------- causal multi-head attention (one wave32 per (b,h,q) row) ----------------
__global__ __launch_bounds__(128) void attn_kernel(
    const float* __restrict__ Q, const float* __restrict__ K,
    const float* __restrict__ V, float* __restrict__ S) {
  __shared__ float attnS[4][Nn];
  int wave = threadIdx.x >> 5, lane = threadIdx.x & 31;
  int gid = blockIdx.x * 4 + wave;
  if (gid >= Bsz * NH * Nn) return;
  int qi = gid % Nn; int bh = gid / Nn; int h = bh % NH; int b = bh / NH;
  const float* qrow = Q + ((size_t)b * Nn + qi) * Dd + h * HD;
  float qreg[HD];
#pragma unroll
  for (int j = 0; j < HD; ++j) qreg[j] = qrow[j];
  float sc[7];
  float mx = -1e30f;
  int nsc = 0;
  for (int c = lane; c <= qi; c += 32) {
    const float* krow = K + ((size_t)b * Nn + c) * Dd + h * HD;
    float acc = 0.f;
#pragma unroll
    for (int j = 0; j < HD; ++j) acc += qreg[j] * krow[j];
    acc *= 0.15811388300841898f;   // 1/sqrt(40)
    sc[nsc++] = acc;
    mx = fmaxf(mx, acc);
  }
  for (int off = 16; off; off >>= 1) mx = fmaxf(mx, __shfl_xor(mx, off, 32));
  float sum = 0.f;
  nsc = 0;
  for (int c = lane; c <= qi; c += 32) { float e = expf(sc[nsc] - mx); sc[nsc++] = e; sum += e; }
  for (int off = 16; off; off >>= 1) sum += __shfl_xor(sum, off, 32);
  float inv = 1.f / sum;
  nsc = 0;
  for (int c = lane; c <= qi; c += 32) attnS[wave][c] = sc[nsc++] * inv;
  // per-wave LDS region; DS ops are in-order within a wave -> no barrier needed
  for (int j = lane; j < HD; j += 32) {
    float acc = 0.f;
    for (int c = 0; c <= qi; ++c)
      acc += attnS[wave][c] * V[((size_t)b * Nn + c) * Dd + h * HD + j];
    S[((size_t)b * Nn + qi) * Dd + h * HD + j] = acc;
  }
}

// ---------------- layernorm over sum of 2 or 4 streams ----------------
__global__ void ln_add_kernel(const float* __restrict__ x0, const float* __restrict__ x1,
                              const float* __restrict__ x2, const float* __restrict__ x3,
                              const float* __restrict__ g, const float* __restrict__ bt,
                              float* __restrict__ out, int rows) {
  int rI = blockIdx.x * blockDim.x + threadIdx.x;
  if (rI >= rows) return;
  size_t base = (size_t)rI * Dd;
  float sum = 0.f, sumsq = 0.f;
  for (int d = 0; d < Dd; ++d) {
    float s = x0[base + d] + x1[base + d];
    if (x2) s += x2[base + d] + x3[base + d];
    sum += s; sumsq += s * s;
  }
  float mu = sum / Dd;
  float var = sumsq / Dd - mu * mu;
  float rs = rsqrtf(var + 1e-5f);
  for (int d = 0; d < Dd; ++d) {
    float s = x0[base + d] + x1[base + d];
    if (x2) s += x2[base + d] + x3[base + d];
    out[base + d] = (s - mu) * rs * g[d] + bt[d];
  }
}

// ---------------- final probability head ----------------
__global__ void prob_kernel(const float* __restrict__ F, const float* __restrict__ pW,
                            const float* __restrict__ pb, float* __restrict__ prob, int rows) {
  int rI = blockIdx.x * blockDim.x + threadIdx.x;
  if (rI >= rows) return;
  const float* fr = F + (size_t)rI * Dd;
  float acc = pb[0];
  for (int d = 0; d < Dd; ++d) acc += fr[d] * pW[d];
  prob[rI] = 1.f / (1.f + expf(-acc));
}

// ===================================================================
extern "C" void kernel_launch(void* const* d_in, const int* in_sizes, int n_in,
                              void* d_out, int out_size, void* d_ws, size_t ws_size,
                              hipStream_t stream) {
  // ---- inputs in setup_inputs() dict order (params flattened in order) ----
  const int*   q_in    = (const int*)  d_in[0];
  const int*   r_in    = (const int*)  d_in[1];
  const float* em_at   = (const float*)d_in[2];
  const float* k_emb   = (const float*)d_in[3];
  const float* v_emb   = (const float*)d_in[4];
  const float* Mk      = (const float*)d_in[5];
  const float* Mv0     = (const float*)d_in[6];
  const float* wb_W    = (const float*)d_in[7];
  const float* wb_b    = (const float*)d_in[8];
  const float* vf_W    = (const float*)d_in[9];
  const float* vf_b    = (const float*)d_in[10];
  const float* e_W     = (const float*)d_in[11];
  const float* e_b     = (const float*)d_in[12];
  const float* a_W     = (const float*)d_in[13];
  const float* a_b     = (const float*)d_in[14];
  const float* f_W     = (const float*)d_in[15];
  const float* f_b     = (const float*)d_in[16];
  const float* P       = (const float*)d_in[17];
  const float* wq_W    = (const float*)d_in[18];
  const float* wq_b    = (const float*)d_in[19];
  const float* wk_W    = (const float*)d_in[20];
  const float* wk_b    = (const float*)d_in[21];
  const float* wv_W    = (const float*)d_in[22];
  const float* wv_b    = (const float*)d_in[23];
  const float* inW     = (const float*)d_in[24];  // [600,200]
  const float* inB     = (const float*)d_in[25];  // [600]
  const float* out_W   = (const float*)d_in[26];
  const float* out_b   = (const float*)d_in[27];
  const float* an_g    = (const float*)d_in[28];
  const float* an_b    = (const float*)d_in[29];
  const float* ffn_W1  = (const float*)d_in[30];
  const float* ffn_b1  = (const float*)d_in[31];
  const float* ffn_W2  = (const float*)d_in[32];
  const float* ffn_b2  = (const float*)d_in[33];
  const float* fn_g    = (const float*)d_in[34];
  const float* fn_b    = (const float*)d_in[35];
  const float* p_W     = (const float*)d_in[36];
  const float* p_b     = (const float*)d_in[37];

  float* prob_out = (float*)d_out;       // [B*N]
  float* Mv_out   = prob_out + BN;       // [B, N+1, SM, D]

  // ---- workspace bump allocator ----
  char* ws = (char*)d_ws;
  size_t off = 0;
  auto alloc = [&](size_t bytes) -> char* {
    char* p = ws + off;
    off = (off + bytes + 255) & ~(size_t)255;
    return p;
  };
  const size_t FB = (size_t)BN * Dd * sizeof(float);   // 10.24 MB
  float* Bf[8];
  for (int i = 0; i < 8; ++i) Bf[i] = (float*)alloc(FB);
  float*     Wsm = (float*)alloc((size_t)BN * SM * sizeof(float));
  _Float16*  EMH = (_Float16*)alloc((size_t)BN * KP768 * sizeof(_Float16));
  _Float16*  AH  = (_Float16*)alloc((size_t)BN * KP400 * sizeof(_Float16));
  auto allocH = [&](int rowsPad, int Kpad) -> _Float16* {
    return (_Float16*)alloc((size_t)rowsPad * Kpad * sizeof(_Float16));
  };
  _Float16* H_wb  = allocH(NP200, KP768);
  _Float16* H_vf  = allocH(NP200, KP400);
  _Float16* H_mk  = allocH(SM,    KP200);
  _Float16* H_e   = allocH(NP200, KP200);
  _Float16* H_a   = allocH(NP200, KP200);
  _Float16* H_f   = allocH(NP200, KP400);
  _Float16* H_wq  = allocH(NP200, KP200);
  _Float16* H_wk  = allocH(NP200, KP200);
  _Float16* H_wv  = allocH(NP200, KP200);
  _Float16* H_iq  = allocH(NP200, KP200);
  _Float16* H_ik  = allocH(NP200, KP200);
  _Float16* H_iv  = allocH(NP200, KP200);
  _Float16* H_ow  = allocH(NP200, KP200);
  _Float16* H_f1  = allocH(NP200, KP200);
  _Float16* H_f2  = allocH(NP200, KP200);
  (void)ws_size; (void)in_sizes; (void)n_in; (void)out_size;

  auto cvt = [&](const float* src, _Float16* dst, int rows, int K, int rowsPad, int Kpad) {
    int total = rowsPad * Kpad;
    cvt_pad_kernel<<<(total + 255) / 256, 256, 0, stream>>>(src, dst, rows, K, rowsPad, Kpad);
  };
  auto cvt2 = [&](const float* s0, const float* s1, _Float16* dst, int K0, int K1, int Kpad) {
    int total = BN * Kpad;
    cvt_pad2_kernel<<<(total + 255) / 256, 256, 0, stream>>>(s0, s1, dst, BN, K0, K1, Kpad);
  };
  auto gemm = [&](const _Float16* A, const _Float16* W, const float* bias, const float* Padd,
                  float* C, int Nc, int NcPad, int Kpad, int act) {
    dim3 g(BN / 16, (NcPad + 63) / 64);
    gemm_wmma_kernel<<<g, 128, 0, stream>>>(A, W, bias, Padd, C, BN, Nc, NcPad, Kpad, act, Nn);
  };

  // ---- convert all weights to padded f16 (small; sits in L2) ----
  cvt(wb_W,  H_wb, Dd, BD,     NP200, KP768);
  cvt(vf_W,  H_vf, Dd, 2 * Dd, NP200, KP400);
  cvt(Mk,    H_mk, SM, Dd,     SM,    KP200);
  cvt(e_W,   H_e,  Dd, Dd,     NP200, KP200);
  cvt(a_W,   H_a,  Dd, Dd,     NP200, KP200);
  cvt(f_W,   H_f,  Dd, 2 * Dd, NP200, KP400);
  cvt(wq_W,  H_wq, Dd, Dd,     NP200, KP200);
  cvt(wk_W,  H_wk, Dd, Dd,     NP200, KP200);
  cvt(wv_W,  H_wv, Dd, Dd,     NP200, KP200);
  cvt(inW,            H_iq, Dd, Dd, NP200, KP200);
  cvt(inW + Dd * Dd,  H_ik, Dd, Dd, NP200, KP200);
  cvt(inW + 2*Dd*Dd,  H_iv, Dd, Dd, NP200, KP200);
  cvt(out_W, H_ow, Dd, Dd,     NP200, KP200);
  cvt(ffn_W1,H_f1, Dd, Dd,     NP200, KP200);
  cvt(ffn_W2,H_f2, Dd, Dd,     NP200, KP200);

  // ---- gathers: k -> B0, v_raw -> B2 ----
  gather_kernel<<<(BN * Dd + 255) / 256, 256, 0, stream>>>(k_emb, v_emb, q_in, r_in, Bf[0], Bf[2]);

  // ---- em = em_at @ wb_W^T + wb_b -> B1 ----
  cvt(em_at, EMH, BN, BD, BN, KP768);
  gemm(EMH, H_wb, wb_b, nullptr, Bf[1], Dd, NP200, KP768, 0);

  // ---- v = concat(em, v_raw) @ vf_W^T + vf_b -> B3 ----
  cvt2(Bf[1], Bf[2], AH, Dd, Dd, KP400);
  gemm(AH, H_vf, vf_b, nullptr, Bf[3], Dd, NP200, KP400, 0);

  // ---- w = softmax(k @ Mk^T) -> Wsm ----
  cvt(Bf[0], AH, BN, Dd, BN, KP200);
  gemm(AH, H_mk, nullptr, nullptr, Wsm, SM, SM, KP200, 0);
  softmax_rows_kernel<<<(BN + 255) / 256, 256, 0, stream>>>(Wsm, BN, SM);

  // ---- e = sigmoid(v@e_W^T+b) -> B4 ; a = tanh(v@a_W^T+b) -> B5 ;
  //      vk -> B6 ; vv -> B7  (all share converted v in AH) ----
  cvt(Bf[3], AH, BN, Dd, BN, KP200);
  gemm(AH, H_e,  e_b,  nullptr, Bf[4], Dd, NP200, KP200, 1);
  gemm(AH, H_a,  a_b,  nullptr, Bf[5], Dd, NP200, KP200, 2);
  gemm(AH, H_wk, wk_b, nullptr, Bf[6], Dd, NP200, KP200, 0);
  gemm(AH, H_wv, wv_b, nullptr, Bf[7], Dd, NP200, KP200, 0);

  // ---- sequential memory recurrence: read -> B1, Mv history -> d_out ----
  recurrence_kernel<<<Bsz, 256, 0, stream>>>(Mv0, Wsm, Bf[4], Bf[5], Bf[1], Mv_out);

  // ---- f = tanh(concat(read, k) @ f_W^T + b) -> B2 ----
  cvt2(Bf[1], Bf[0], AH, Dd, Dd, KP400);
  gemm(AH, H_f, f_b, nullptr, Bf[2], Dd, NP200, KP400, 2);

  // ---- vq = f @ wq^T + b + P -> B4 ----
  cvt(Bf[2], AH, BN, Dd, BN, KP200);
  gemm(AH, H_wq, wq_b, P, Bf[4], Dd, NP200, KP200, 0);

  // ---- in-proj: Q -> B5, K -> B0, V -> B1 ----
  cvt(Bf[4], AH, BN, Dd, BN, KP200);
  gemm(AH, H_iq, inB, nullptr, Bf[5], Dd, NP200, KP200, 0);
  cvt(Bf[6], AH, BN, Dd, BN, KP200);
  gemm(AH, H_ik, inB + Dd, nullptr, Bf[0], Dd, NP200, KP200, 0);
  cvt(Bf[7], AH, BN, Dd, BN, KP200);
  gemm(AH, H_iv, inB + 2 * Dd, nullptr, Bf[1], Dd, NP200, KP200, 0);

  // ---- causal MHA -> B2 ----
  attn_kernel<<<(Bsz * NH * Nn + 3) / 4, 128, 0, stream>>>(Bf[5], Bf[0], Bf[1], Bf[2]);

  // ---- out proj -> B3 ----
  cvt(Bf[2], AH, BN, Dd, BN, KP200);
  gemm(AH, H_ow, out_b, nullptr, Bf[3], Dd, NP200, KP200, 0);

  // ---- X = LN(Sproj + vq + vk + vv) -> B0 ----
  ln_add_kernel<<<(BN + 255) / 256, 256, 0, stream>>>(Bf[3], Bf[4], Bf[6], Bf[7],
                                                      an_g, an_b, Bf[0], BN);

  // ---- FFN: relu(X@W1+b1) -> B1 ; @W2+b2 -> B2 ----
  cvt(Bf[0], AH, BN, Dd, BN, KP200);
  gemm(AH, H_f1, ffn_b1, nullptr, Bf[1], Dd, NP200, KP200, 3);
  cvt(Bf[1], AH, BN, Dd, BN, KP200);
  gemm(AH, H_f2, ffn_b2, nullptr, Bf[2], Dd, NP200, KP200, 0);

  // ---- F = LN(ffn2 + X) -> B3 ----
  ln_add_kernel<<<(BN + 255) / 256, 256, 0, stream>>>(Bf[2], Bf[0], nullptr, nullptr,
                                                      fn_g, fn_b, Bf[3], BN);

  // ---- prob = sigmoid(F @ p_W^T + p_b) -> d_out[0:BN] ----
  prob_kernel<<<(BN + 255) / 256, 256, 0, stream>>>(Bf[3], p_W, p_b, prob_out, BN);
}
